// BaseNet_5918464934141
// MI455X (gfx1250) — compile-verified, matrix operand order checked by reading, not run on hardware
//
#include <hip/hip_runtime.h>
#include <hip/hip_bf16.h>

// ---------------------------------------------------------------------------
// GRU scan for MI455X (gfx1250, wave32, WMMA, async global->LDS).
//
// Phase A (parallel):  gi[t][wg][b][g*16+j] = (x @ w_ih^T + b_ih), bf16,
//                      blocked so each scan workgroup reads contiguous 6KB/step.
// Phase B (scan):      32 persistent workgroups, one per 16 hidden columns.
//                      w_hh gate slices resident in LDS for all 2048 steps.
//                      Per step: async-copy bf16 h + gi block into LDS,
//                      16x v_wmma_f32_16x16x32_bf16 per gate, fp32 gate math
//                      (masks folded into the epilogue: (m*h)@W = m*(h@W)),
//                      global barrier, next step. fp32 hidden carry kept.
// Fallback (ws too small): fused variant recomputes the x-projection per step.
// ---------------------------------------------------------------------------

#define T_STEPS 2048
#define NB      64          // batch
#define II      512         // input dim
#define HH      512         // hidden dim
#define NWG     32          // scan workgroups; each owns 16 hidden columns
#define LDSP    520         // padded K-stride for 512-wide LDS tiles (16B align)

#define WHH_E   (3 * 16 * LDSP)   // 24960 ushorts
#define TILE_E  (64 * LDSP)       // 33280 ushorts
#define GIS_E   (64 * 48)         // 3072 ushorts

#if defined(__has_builtin)
#if __has_builtin(__builtin_amdgcn_global_load_async_to_lds_b128) && \
    __has_builtin(__builtin_amdgcn_s_wait_asynccnt)
#define HAVE_ASYNC 1
#endif
#endif
#ifndef HAVE_ASYNC
#define HAVE_ASYNC 0
#endif

typedef __attribute__((ext_vector_type(16))) __bf16 v16bf;
typedef __attribute__((ext_vector_type(8)))  float  v8f;

union FragU { uint4 q[2]; v16bf v; };

__device__ __forceinline__ unsigned short f2bf(float f) {
  unsigned u = __builtin_bit_cast(unsigned, f);
  u += 0x7FFFu + ((u >> 16) & 1u);           // round-to-nearest-even
  return (unsigned short)(u >> 16);
}
__device__ __forceinline__ float bf2f(unsigned short u) {
  return __builtin_bit_cast(float, (unsigned)u << 16);
}
__device__ __forceinline__ float sigm(float x) {
  return 1.0f / (1.0f + __expf(-x));
}

#if HAVE_ASYNC
// Builtin signature (from clang diagnostic): AS1 int4* src, AS3 int4* dst,
// imm offset, imm cpol.  ("__device__" in the diag == addrspace(1) in HIP.)
typedef int async_v4i __attribute__((vector_size(16)));
#define ASYNC_CP16(gp, lp)                                                   \
  __builtin_amdgcn_global_load_async_to_lds_b128(                            \
      (__attribute__((address_space(1))) async_v4i*)(gp),                    \
      (__attribute__((address_space(3))) async_v4i*)(lp), 0, 0)
#endif

// Load a 16x32 bf16 fragment (A-layout, ISA 7.12.2) from an LDS tile.
// lane 0-15: rows rowBase..+15, K = {0..7,16..23}; lane 16-31: K = {8..15,24..31}
// Two 16B-contiguous halves -> two ds_load_b128.
__device__ __forceinline__ v16bf frag_ld(const unsigned short* tile, int rowBase,
                                         int stride, int kbase, int lane) {
  const int r = rowBase + (lane & 15);
  const int k = kbase + ((lane >> 4) << 3);
  const unsigned short* p = tile + r * stride + k;
  FragU f;
  f.q[0] = *(const uint4*)(p);
  f.q[1] = *(const uint4*)(p + 16);
  return f.v;
}

__device__ __forceinline__ v8f wmma_bf16(v16bf a, v16bf b, v8f c) {
  return __builtin_amdgcn_wmma_f32_16x16x32_bf16(false, a, false, b,
                                                 (short)0, c, false, false);
}

// Convert-stage a [64][512] fp32 matrix into a bf16 LDS tile (LDSP stride).
__device__ __forceinline__ void stage_f32(const float* __restrict__ src,
                                          unsigned short* __restrict__ dst,
                                          int tid, int nthr) {
  for (int i = tid; i < (NB * HH) / 4; i += nthr) {
    const int b = i >> 7, k4 = (i & 127) << 2;
    const float4 f = *(const float4*)(src + (size_t)b * HH + k4);
    uint2 p;
    p.x = (unsigned)f2bf(f.x) | ((unsigned)f2bf(f.y) << 16);
    p.y = (unsigned)f2bf(f.z) | ((unsigned)f2bf(f.w) << 16);
    *(uint2*)(dst + b * LDSP + k4) = p;
  }
}

// ---------------------------------------------------------------------------
// Phase A: gi = x @ w_ih^T + b_ih  (bf16 out, blocked [T][NWG][64][48])
// grid = (2048, 8); block = 256 (8 waves). Each WG: one t, 192 gate columns.
// Index decompositions are division-free: tile T = (wave>>2)*6 + nt, and
// (wave>>2)*6 is divisible by 3, so wgl/g split is compile-time per nt.
// ---------------------------------------------------------------------------
__global__ __launch_bounds__(256)
void gi_gemm_kernel(const float* __restrict__ x,
                    const float* __restrict__ w_ih,
                    const float* __restrict__ b_ih,
                    unsigned short* __restrict__ gi) {
  const int t    = blockIdx.x;
  const int c    = blockIdx.y;            // gate chunk: 192 columns
  const int tid  = threadIdx.x;
  const int lane = tid & 31;
  const int wave = tid >> 5;
  const int m0   = (wave & 3) * 16;       // 4 waves across batch rows
  const int half = wave >> 2;             // 2 waves across gate columns
  const int q0   = half * 96;
  const int ncol = lane & 15;

  __shared__ __align__(16) unsigned short aT[64 * 40];    // x slab  [64][32]+pad
  __shared__ __align__(16) unsigned short bT[192 * 40];   // w slab [192][32]+pad

  // Per-thread staging addresses, kt-invariant (computed once).
  int arow[2], ak4[2];
#pragma unroll
  for (int j = 0; j < 2; ++j) {
    const int i = tid + j * 256;
    arow[j] = i >> 3;
    ak4[j]  = (i & 7) << 2;
  }
  int brow[6], bq[6], bk4[6];
#pragma unroll
  for (int j = 0; j < 6; ++j) {
    const int i = tid + j * 256;
    const int q = i >> 3;
    const int T16 = q >> 4;               // 0..11
    const int d3  = T16 / 3;              // once per j, not per element
    brow[j] = (T16 - d3 * 3) * HH + (c * 4 + d3) * 16 + (q & 15);
    bq[j]   = q;
    bk4[j]  = (i & 7) << 2;
  }
  const float* xrow = x + (size_t)t * 64 * II;

  v8f acc[6] = {};

  for (int kt = 0; kt < 16; ++kt) {
    const int kb = kt * 32;
#pragma unroll
    for (int j = 0; j < 2; ++j) {
      const float4 f = *(const float4*)(xrow + (size_t)arow[j] * II + kb + ak4[j]);
      uint2 p;
      p.x = (unsigned)f2bf(f.x) | ((unsigned)f2bf(f.y) << 16);
      p.y = (unsigned)f2bf(f.z) | ((unsigned)f2bf(f.w) << 16);
      *(uint2*)(aT + arow[j] * 40 + ak4[j]) = p;
    }
#pragma unroll
    for (int j = 0; j < 6; ++j) {
      const float4 f = *(const float4*)(w_ih + (size_t)brow[j] * II + kb + bk4[j]);
      uint2 p;
      p.x = (unsigned)f2bf(f.x) | ((unsigned)f2bf(f.y) << 16);
      p.y = (unsigned)f2bf(f.z) | ((unsigned)f2bf(f.w) << 16);
      *(uint2*)(bT + bq[j] * 40 + bk4[j]) = p;
    }
    __syncthreads();
    const v16bf a = frag_ld(aT, m0, 40, 0, lane);
#pragma unroll
    for (int nt = 0; nt < 6; ++nt)
      acc[nt] = wmma_bf16(a, frag_ld(bT, q0 + nt * 16, 40, 0, lane), acc[nt]);
    __syncthreads();
  }

#pragma unroll
  for (int nt = 0; nt < 6; ++nt) {
    const int wgl = c * 4 + half * 2 + nt / 3;  // nt/3, nt%3 are compile-time
    const int g   = nt % 3;
    const float bias = b_ih[g * HH + wgl * 16 + ncol];
    const size_t base = (((size_t)t * NWG + wgl) * 64) * 48 + g * 16 + ncol;
#pragma unroll
    for (int v = 0; v < 8; ++v) {
      const int m = m0 + v + ((lane >> 4) << 3);          // C/D layout
      gi[base + (size_t)m * 48] = f2bf(acc[nt][v] + bias);
    }
  }
}

// ---------------------------------------------------------------------------
// Phase B: persistent scan. grid = 32, block = 128 (4 waves).
// WG wg owns hidden columns [wg*16, wg*16+16); waves split the 64 batch rows.
// ---------------------------------------------------------------------------
template <bool PRECOMP>
__global__ __launch_bounds__(128)
void gru_scan_kernel(const float* __restrict__ x,
                     const float* __restrict__ h0,
                     const float* __restrict__ masks,
                     const float* __restrict__ w_ih,
                     const float* __restrict__ w_hh,
                     const float* __restrict__ b_ih,
                     const float* __restrict__ b_hh,
                     const unsigned short* __restrict__ gi,
                     float* __restrict__ out,
                     float* __restrict__ hfinal,
                     float* __restrict__ hbuf32,          // [2][NB*HH] fp32 carry
                     unsigned short* __restrict__ hbuf16, // [2][NB*HH] bf16 mirror
                     unsigned* __restrict__ bar) {        // {count, phase}
  const int wg   = blockIdx.x;
  const int tid  = threadIdx.x;
  const int lane = tid & 31;
  const int wave = tid >> 5;
  const int j0   = wg * 16;
  const int m0   = wave * 16;
  const int ncol = lane & 15;

  extern __shared__ __align__(16) char smem_raw[];
  unsigned short* whh_s = (unsigned short*)smem_raw;      // [3][16][LDSP]
  unsigned short* tileH = whh_s + WHH_E;                  // [64][LDSP] bf16 h
  unsigned short* gis   = tileH + TILE_E;                 // [64][48]  (PRECOMP)
  unsigned short* wih_s = tileH + TILE_E;                 // [3][16][LDSP] (fused)
  unsigned short* tileX = wih_s + WHH_E;                  // [64][LDSP]   (fused)

  // Resident weight slices: loaded once, reused for all 2048 steps.
  for (int i = tid; i < (3 * 16 * II) / 4; i += 128) {
    const int r = i >> 7, k4 = (i & 127) << 2;            // r = g*16+jj
    const int g = r >> 4, jj = r & 15;
    {
      const float4 f = *(const float4*)(w_hh + ((size_t)g * HH + j0 + jj) * HH + k4);
      uint2 p;
      p.x = (unsigned)f2bf(f.x) | ((unsigned)f2bf(f.y) << 16);
      p.y = (unsigned)f2bf(f.z) | ((unsigned)f2bf(f.w) << 16);
      *(uint2*)(whh_s + r * LDSP + k4) = p;
    }
    if (!PRECOMP) {
      const float4 f = *(const float4*)(w_ih + ((size_t)g * HH + j0 + jj) * II + k4);
      uint2 p;
      p.x = (unsigned)f2bf(f.x) | ((unsigned)f2bf(f.y) << 16);
      p.y = (unsigned)f2bf(f.z) | ((unsigned)f2bf(f.w) << 16);
      *(uint2*)(wih_s + r * LDSP + k4) = p;
    }
  }

  // Per-lane biases (column-indexed -> loop invariant).
  const float bhh0 = b_hh[0 * HH + j0 + ncol];
  const float bhh1 = b_hh[1 * HH + j0 + ncol];
  const float bhh2 = b_hh[2 * HH + j0 + ncol];
  float bih0 = 0.f, bih1 = 0.f, bih2 = 0.f;
  if (!PRECOMP) {
    bih0 = b_ih[0 * HH + j0 + ncol];
    bih1 = b_ih[1 * HH + j0 + ncol];
    bih2 = b_ih[2 * HH + j0 + ncol];
  }
  __syncthreads();

  for (int t = 0; t < T_STEPS; ++t) {
    const int cur = t & 1, prv = cur ^ 1;
    const float* hsrc32 = (t == 0) ? h0 : (hbuf32 + (size_t)prv * NB * HH);

    // ---- Stage operands for this step --------------------------------
#if HAVE_ASYNC
    if (t > 0) {
      // Pure bf16 copy (mask folded into epilogue): async DMA, 16B/lane.
      const char* gsrc = (const char*)(hbuf16 + (size_t)prv * NB * HH);
      char* lbase = (char*)tileH;
      for (int c = tid; c < 4096; c += 128) {             // 32 issues/thread
        const int b = c >> 6, off = (c & 63) << 4;
        ASYNC_CP16(gsrc + (size_t)b * 1024 + off, lbase + b * (LDSP * 2) + off);
      }
    } else {
      stage_f32(h0, tileH, tid, 128);
    }
    if (PRECOMP) {  // gi block: issued now, waited on only after the GEMM
      const char* gsrc = (const char*)(gi + (((size_t)t * NWG + wg) * 64) * 48);
      char* lbase = (char*)gis;
      for (int c = tid; c < 384; c += 128)                // 3 issues/thread
        ASYNC_CP16(gsrc + ((size_t)c << 4), lbase + (c << 4));
    }
#else
    stage_f32(hsrc32, tileH, tid, 128);
    if (PRECOMP && t + 1 < T_STEPS) {
      const char* np = (const char*)(gi + (((size_t)(t + 1) * NWG + wg) * 64) * 48);
      __builtin_prefetch(np + tid * 48, 0, 1);
    }
#endif

    v8f accI0 = {}, accI1 = {}, accI2 = {};
    if (!PRECOMP) {  // fused input projection from x[t]
      stage_f32(x + (size_t)t * NB * II, tileX, tid, 128);
    }

#if HAVE_ASYNC
    if (PRECOMP) __builtin_amdgcn_s_wait_asynccnt(3);     // h done, gi in flight
    else         __builtin_amdgcn_s_wait_asynccnt(0);
#endif
    __syncthreads();

    if (!PRECOMP) {
      for (int kt = 0; kt < 16; ++kt) {
        const v16bf a = frag_ld(tileX, m0, LDSP, kt * 32, lane);
        accI0 = wmma_bf16(a, frag_ld(wih_s + 0 * 16 * LDSP, 0, LDSP, kt * 32, lane), accI0);
        accI1 = wmma_bf16(a, frag_ld(wih_s + 1 * 16 * LDSP, 0, LDSP, kt * 32, lane), accI1);
        accI2 = wmma_bf16(a, frag_ld(wih_s + 2 * 16 * LDSP, 0, LDSP, kt * 32, lane), accI2);
      }
    }

    // ---- Recurrent GEMM: gh(unmasked) = h_prev @ w_hh_slice^T ---------
    v8f acc0 = {}, acc1 = {}, acc2 = {};
    for (int kt = 0; kt < 16; ++kt) {
      const v16bf a = frag_ld(tileH, m0, LDSP, kt * 32, lane);
      acc0 = wmma_bf16(a, frag_ld(whh_s + 0 * 16 * LDSP, 0, LDSP, kt * 32, lane), acc0);
      acc1 = wmma_bf16(a, frag_ld(whh_s + 1 * 16 * LDSP, 0, LDSP, kt * 32, lane), acc1);
      acc2 = wmma_bf16(a, frag_ld(whh_s + 2 * 16 * LDSP, 0, LDSP, kt * 32, lane), acc2);
    }

#if HAVE_ASYNC
    if (PRECOMP) {
      __builtin_amdgcn_s_wait_asynccnt(0);                // gi block landed
      __syncthreads();                                    // visible to all waves
    }
#endif

    // ---- Gate math in fp32; mask applied here: (m*h)@W = m*(h@W) ------
    const unsigned short* giB =
        PRECOMP ? (gi + (((size_t)t * NWG + wg) * 64) * 48) : nullptr;
#pragma unroll
    for (int v = 0; v < 8; ++v) {
      const int m = m0 + v + ((lane >> 4) << 3);          // C/D row mapping
      const float mk = masks[(size_t)t * NB + m];
      float ir, iz, in_;
      if (PRECOMP) {
#if HAVE_ASYNC
        ir  = bf2f(gis[m * 48 + 0 * 16 + ncol]);
        iz  = bf2f(gis[m * 48 + 1 * 16 + ncol]);
        in_ = bf2f(gis[m * 48 + 2 * 16 + ncol]);
#else
        ir  = bf2f(giB[m * 48 + 0 * 16 + ncol]);
        iz  = bf2f(giB[m * 48 + 1 * 16 + ncol]);
        in_ = bf2f(giB[m * 48 + 2 * 16 + ncol]);
#endif
      } else {
        ir = accI0[v] + bih0; iz = accI1[v] + bih1; in_ = accI2[v] + bih2;
      }
      const float hr = mk * acc0[v] + bhh0;
      const float hz = mk * acc1[v] + bhh1;
      const float hn = mk * acc2[v] + bhh2;
      const float r  = sigm(ir + hr);
      const float z  = sigm(iz + hz);
      const float nn = tanhf(in_ + r * hn);
      const float hp = mk * hsrc32[(size_t)m * HH + j0 + ncol];
      const float hnew = (1.0f - z) * nn + z * hp;
      out[((size_t)t * NB + m) * HH + j0 + ncol] = hnew;
      hbuf32[(size_t)cur * NB * HH + (size_t)m * HH + j0 + ncol] = hnew;
#if HAVE_ASYNC
      hbuf16[(size_t)cur * NB * HH + (size_t)m * HH + j0 + ncol] = f2bf(hnew);
#endif
    }

    // ---- Grid-wide barrier (one per timestep) -------------------------
    __threadfence();   // device-scope release of this wave's h stores
    __syncthreads();
    if (tid == 0) {
      volatile unsigned* vph = bar + 1;
      const unsigned ph = *vph;
      if (atomicAdd(bar, 1u) == NWG - 1) {
        *(volatile unsigned*)bar = 0u;
        __threadfence();
        atomicAdd(bar + 1, 1u);
      } else {
        while (*vph == ph) __builtin_amdgcn_s_sleep(1);
      }
    }
    __syncthreads();
    // agent-scope acquire: invalidate stale WGP$ lines before reading the
    // other workgroups' h (regular loads and async-DMA reads next step).
    __builtin_amdgcn_fence(__ATOMIC_ACQUIRE, "agent");
  }

  // h_final = h after the last step (this WG's 16 columns).
  const float* hlast = hbuf32 + (size_t)((T_STEPS - 1) & 1) * NB * HH;
  for (int i = tid; i < NB * 16; i += 128) {
    const int b = i >> 4, d = i & 15;
    hfinal[(size_t)b * HH + j0 + d] = hlast[(size_t)b * HH + j0 + d];
  }
}

// ---------------------------------------------------------------------------
extern "C" void kernel_launch(void* const* d_in, const int* in_sizes, int n_in,
                              void* d_out, int out_size, void* d_ws, size_t ws_size,
                              hipStream_t stream) {
  const float* x     = (const float*)d_in[0];
  const float* h0    = (const float*)d_in[1];
  const float* masks = (const float*)d_in[2];
  const float* w_ih  = (const float*)d_in[3];
  const float* w_hh  = (const float*)d_in[4];
  const float* b_ih  = (const float*)d_in[5];
  const float* b_hh  = (const float*)d_in[6];

  float* out    = (float*)d_out;                       // [T*N, H]
  float* hfinal = out + (size_t)T_STEPS * NB * HH;     // [N, H]

  const size_t GI_BYTES  = (size_t)T_STEPS * NWG * 64 * 48 * 2;  // ~403MB bf16
  const size_t H32_BYTES = (size_t)2 * NB * HH * sizeof(float);
  const size_t H16_BYTES = (size_t)2 * NB * HH * sizeof(unsigned short);
  const bool precomp = ws_size >= GI_BYTES + H32_BYTES + H16_BYTES + 256;

  char* wsc = (char*)d_ws;
  unsigned short* gi = nullptr;
  size_t off = 0;
  if (precomp) { gi = (unsigned short*)wsc; off += GI_BYTES; }
  float* hbuf32 = (float*)(wsc + off);          off += H32_BYTES;
  unsigned short* hbuf16 = (unsigned short*)(wsc + off); off += H16_BYTES;
  unsigned* bar = (unsigned*)(wsc + off);
  (void)hipMemsetAsync(bar, 0, 64, stream);  // barrier state starts at 0 each call

  if (precomp) {
    dim3 grid(T_STEPS, 8);
    gi_gemm_kernel<<<grid, 256, 0, stream>>>(x, w_ih, b_ih, gi);

    const size_t lds = (size_t)(WHH_E + TILE_E + GIS_E) * 2;
    (void)hipFuncSetAttribute((const void*)&gru_scan_kernel<true>,
                              hipFuncAttributeMaxDynamicSharedMemorySize, (int)lds);
    gru_scan_kernel<true><<<NWG, 128, lds, stream>>>(
        x, h0, masks, w_ih, w_hh, b_ih, b_hh, gi, out, hfinal,
        hbuf32, hbuf16, bar);
  } else {
    const size_t lds = (size_t)(2 * WHH_E + 2 * TILE_E) * 2;
    (void)hipFuncSetAttribute((const void*)&gru_scan_kernel<false>,
                              hipFuncAttributeMaxDynamicSharedMemorySize, (int)lds);
    gru_scan_kernel<false><<<NWG, 128, lds, stream>>>(
        x, h0, masks, w_ih, w_hh, b_ih, b_hh, nullptr, out, hfinal,
        hbuf32, hbuf16, bar);
  }
}
